// MultiHeadAttention_76141180223583
// MI455X (gfx1250) — matrix operand order repaired and checked by reading, not compiled, not run on hardware
//
#include <hip/hip_runtime.h>
#include <hip/hip_bf16.h>
#include <math.h>

#define S_LEN  2048
#define BATCH  2
#define DMODEL 1024
#define NHEAD  16
#define DHEAD  64

typedef __attribute__((ext_vector_type(16))) __bf16 v16bf;
typedef __attribute__((ext_vector_type(8)))  float  v8f;

// ---------------------------------------------------------------------------
// WMMA fragment loaders (CDNA5 16-bit layouts, CDNA5 ISA §7.12.2)
// A 16x32: lane<16 holds row (lane&15), elems 0..7 = K[kh..kh+7],
//          elems 8..15 = K[kh+16..kh+23], kh = (lane>>4)*8
// B 32x16: lane<16 holds col (lane&15), elems 0..7 = K[kb..kb+7],
//          elems 8..15 = K[kb+16..kb+23], kb = (lane>>4)*8 (same as A)
// ---------------------------------------------------------------------------
__device__ __forceinline__ v16bf load_a_frag(const __bf16* p) {
  union { uint4 u[2]; v16bf v; } t;
  t.u[0] = *(const uint4*)(p);
  t.u[1] = *(const uint4*)(p + 16);
  return t.v;
}
__device__ __forceinline__ v16bf load_b_frag(const __bf16* p) {
  union { uint4 u[2]; v16bf v; } t;
  t.u[0] = *(const uint4*)(p);
  t.u[1] = *(const uint4*)(p + 16);
  return t.v;
}

// ---------------------------------------------------------------------------
// Generic NT GEMM: C[M,N] = A[M,K] * B[N,K]^T  (A,B bf16 row-major)
// Optional: per-column fp32 bias, Transformer-XL relative-shift additive term
// (posAdd, ld=Spos), global scale; writes fp32 Cf and/or bf16 Cb.
// Block = 256 thr (8 waves), tile 128x128, waves on a 2(M)x4(N) grid,
// each wave: 4x2 v_wmma_f32_16x16x32_bf16 accumulators.
// Double-buffered LDS pipeline: tile k+1 staged in VGPRs while WMMAs consume
// tile k from LDS; global_prefetch two K-steps ahead.
// ---------------------------------------------------------------------------
__global__ __launch_bounds__(256) void gemm_nt_kernel(
    const __bf16* __restrict__ A, int lda,
    const __bf16* __restrict__ Bm, int ldb,
    int M, int N, int K,
    const float* __restrict__ bias,
    const float* __restrict__ posAdd, int Spos,
    float scale,
    float* __restrict__ Cf, __bf16* __restrict__ Cb, int ldc)
{
  __shared__ __align__(16) __bf16 At[2][128 * 40];   // 128 rows x 32 K, pad->40
  __shared__ __align__(16) __bf16 Bt[2][128 * 40];   // 128 cols x 32 K, pad->40

  const int tid  = threadIdx.x;
  const int r0   = blockIdx.y * 128;
  const int c0   = blockIdx.x * 128;
  const int w    = tid >> 5, lane = tid & 31;
  const int waveM = (w >> 2) * 64;     // 0 / 64
  const int waveN = (w & 3) * 32;      // 0 / 32 / 64 / 96
  const int lr = lane & 15, lh = lane >> 4;

  v8f acc[4][2];
#pragma unroll
  for (int mt = 0; mt < 4; ++mt)
#pragma unroll
    for (int nt = 0; nt < 2; ++nt)
#pragma unroll
      for (int i = 0; i < 8; ++i) acc[mt][nt][i] = 0.0f;

  // tile load mapping: thread -> (row tid>>1, 16-wide half tid&1)
  const int trow  = tid >> 1;
  const int thalf = (tid & 1) * 16;
  int ga = r0 + trow; if (ga > M - 1) ga = M - 1;   // clamp; masked at store
  int gb = c0 + trow; if (gb > N - 1) gb = N - 1;
  const __bf16* gA = A  + (size_t)ga * lda + thalf;
  const __bf16* gB = Bm + (size_t)gb * ldb + thalf;

  const int niter = K >> 5;            // K is a multiple of 32 in all uses
  uint4 ra0, ra1, rb0, rb1;            // staged tile (registers)

  // prologue: stage tile 0
  {
    const __bf16* sA = gA;
    ra0 = *(const uint4*)sA; ra1 = *(const uint4*)(sA + 8);
    const __bf16* sB = gB;
    rb0 = *(const uint4*)sB; rb1 = *(const uint4*)(sB + 8);
  }

  for (int it = 0; it < niter; ++it) {
    const int buf = it & 1;
    // commit staged tile to LDS
    {
      __bf16* dA = &At[buf][trow * 40 + thalf];
      *(uint4*)dA = ra0; *(uint4*)(dA + 8) = ra1;
      __bf16* dB = &Bt[buf][trow * 40 + thalf];
      *(uint4*)dB = rb0; *(uint4*)(dB + 8) = rb1;
    }
    __syncthreads();

    // stage tile it+1 (loads in flight during the WMMA phase below)
    if (it + 1 < niter) {
      const __bf16* sA = gA + ((it + 1) << 5);
      ra0 = *(const uint4*)sA; ra1 = *(const uint4*)(sA + 8);
      const __bf16* sB = gB + ((it + 1) << 5);
      rb0 = *(const uint4*)sB; rb1 = *(const uint4*)(sB + 8);
    }
    // warm caches two K-steps ahead (global_prefetch_b8)
    if (it + 2 < niter) {
      __builtin_prefetch(gA + ((it + 2) << 5), 0, 0);
      __builtin_prefetch(gB + ((it + 2) << 5), 0, 0);
    }

    v16bf af[4], bfv[2];
#pragma unroll
    for (int mt = 0; mt < 4; ++mt)
      af[mt] = load_a_frag(&At[buf][(waveM + mt * 16 + lr) * 40 + lh * 8]);
#pragma unroll
    for (int nt = 0; nt < 2; ++nt)
      bfv[nt] = load_b_frag(&Bt[buf][(waveN + nt * 16 + lr) * 40 + lh * 8]);

#pragma unroll
    for (int mt = 0; mt < 4; ++mt)
#pragma unroll
      for (int nt = 0; nt < 2; ++nt)
        acc[mt][nt] = __builtin_amdgcn_wmma_f32_16x16x32_bf16(
            false, af[mt], false, bfv[nt], (short)0, acc[mt][nt], false, false);
    __syncthreads();
  }

  // Epilogue. D-fragment layout: elem i of v8f -> M = base + i (+8 for upper
  // lane half), N = lane&15.
#pragma unroll
  for (int mt = 0; mt < 4; ++mt) {
#pragma unroll
    for (int nt = 0; nt < 2; ++nt) {
      int n = c0 + waveN + nt * 16 + lr;
      if (n >= N) continue;
      int mbase = r0 + waveM + mt * 16 + lh * 8;
      float bv = bias ? bias[n] : 0.0f;
#pragma unroll
      for (int i = 0; i < 8; ++i) {
        int m = mbase + i;
        if (m >= M) continue;
        float vv = acc[mt][nt][i] + bv;
        if (posAdd) {
          // exact _relative_shift semantics:
          //   c<=r : pos[r, S-1-r+c];  c==r+1 : 0;  c>=r+2 : pos[r+1, c-r-2]
          float pv;
          if (n <= m)          pv = posAdd[(size_t)m * Spos + (Spos - 1 - m + n)];
          else if (n == m + 1) pv = 0.0f;
          else                 pv = posAdd[(size_t)(m + 1) * Spos + (n - m - 2)];
          vv += pv;
        }
        vv *= scale;
        if (Cf) Cf[(size_t)m * ldc + n] = vv;
        if (Cb) Cb[(size_t)m * ldc + n] = (__bf16)vv;
      }
    }
  }
}

// ---------------------------------------------------------------------------
// block reductions (256 threads = 8 wave32)
// ---------------------------------------------------------------------------
__device__ __forceinline__ float block_sum(float v) {
  __shared__ float red[8];
#pragma unroll
  for (int o = 16; o > 0; o >>= 1) v += __shfl_xor(v, o, 32);
  int w = threadIdx.x >> 5, lane = threadIdx.x & 31;
  __syncthreads();
  if (lane == 0) red[w] = v;
  __syncthreads();
  float s = red[0];
#pragma unroll
  for (int i = 1; i < 8; ++i) s += red[i];
  return s;
}
__device__ __forceinline__ float block_max(float v) {
  __shared__ float redm[8];
#pragma unroll
  for (int o = 16; o > 0; o >>= 1) v = fmaxf(v, __shfl_xor(v, o, 32));
  int w = threadIdx.x >> 5, lane = threadIdx.x & 31;
  __syncthreads();
  if (lane == 0) redm[w] = v;
  __syncthreads();
  float s = redm[0];
#pragma unroll
  for (int i = 1; i < 8; ++i) s = fmaxf(s, redm[i]);
  return s;
}

// LayerNorm row (D=1024) -> bf16
__global__ __launch_bounds__(256) void layernorm_kernel(
    const float* __restrict__ x, const float* __restrict__ g,
    const float* __restrict__ b, __bf16* __restrict__ out)
{
  const int row = blockIdx.x, tid = threadIdx.x;
  const size_t base = (size_t)row * DMODEL;
  float v[4]; float s = 0.0f;
#pragma unroll
  for (int i = 0; i < 4; ++i) { v[i] = x[base + tid + i * 256]; s += v[i]; }
  float mu = block_sum(s) * (1.0f / DMODEL);
  float q = 0.0f;
#pragma unroll
  for (int i = 0; i < 4; ++i) { float d = v[i] - mu; q += d * d; }
  float rstd = rsqrtf(block_sum(q) * (1.0f / DMODEL) + 1e-5f);
#pragma unroll
  for (int i = 0; i < 4; ++i) {
    int c = tid + i * 256;
    out[base + c] = (__bf16)((v[i] - mu) * rstd * g[c] + b[c]);
  }
}

// sinusoidal positional encoding -> bf16 [S, D]
__global__ __launch_bounds__(256) void pe_kernel(__bf16* __restrict__ pe) {
  const int srow = blockIdx.x, tid = threadIdx.x;
  const float c = -9.210340371976184f / (float)DMODEL;  // -ln(10000)/D
#pragma unroll
  for (int j = 0; j < 2; ++j) {
    int p = tid + j * 256;                 // pair index 0..511
    float dv = __expf((2.0f * p) * c);
    float ang = (float)srow * dv;
    float sn, cs;
    __sincosf(ang, &sn, &cs);
    pe[(size_t)srow * DMODEL + 2 * p]     = (__bf16)sn;
    pe[(size_t)srow * DMODEL + 2 * p + 1] = (__bf16)cs;
  }
}

// W [K,N] fp32 -> Wt [N,K] bf16 (so GEMM can consume NT form)
__global__ void transpose_w_kernel(const float* __restrict__ W,
                                   __bf16* __restrict__ Wt, int K, int N) {
  int idx = blockIdx.x * 256 + threadIdx.x;
  if (idx < K * N) {
    int k = idx / N, n = idx - k * N;
    Wt[(size_t)n * K + k] = (__bf16)W[idx];
  }
}

// bqu = bq + u_flat ; bqv = bq + v_flat   (u,v are [H,DH] == flat D columns)
__global__ void make_bias_kernel(const float* __restrict__ bq,
                                 const float* __restrict__ u,
                                 const float* __restrict__ v,
                                 float* __restrict__ bqu, float* __restrict__ bqv) {
  int i = blockIdx.x * 256 + threadIdx.x;
  if (i < DMODEL) { bqu[i] = bq[i] + u[i]; bqv[i] = bq[i] + v[i]; }
}

// V bf16 [B*S, D] -> vt bf16 [B][H][DH][S]  (NT form for attn @ V)
__global__ void vtrans_kernel(const __bf16* __restrict__ Vb,
                              __bf16* __restrict__ vt) {
  int idx = blockIdx.x * 256 + threadIdx.x;
  if (idx < BATCH * S_LEN * DMODEL) {
    int b = idx / (S_LEN * DMODEL);
    int r = idx - b * (S_LEN * DMODEL);
    int j = r / DMODEL;
    int cc = r - j * DMODEL;
    int h = cc >> 6, d = cc & 63;
    vt[((size_t)((b * NHEAD + h) * DHEAD + d)) * S_LEN + j] = Vb[idx];
  }
}

// row softmax over S=2048, fp32 in -> bf16 out
__global__ __launch_bounds__(256) void softmax_kernel(
    const float* __restrict__ sc, __bf16* __restrict__ out) {
  const int row = blockIdx.x, tid = threadIdx.x;
  const size_t base = (size_t)row * S_LEN;
  float v[8]; float mx = -3.4e38f;
#pragma unroll
  for (int i = 0; i < 8; ++i) { v[i] = sc[base + tid + i * 256]; mx = fmaxf(mx, v[i]); }
  mx = block_max(mx);
  float s = 0.0f;
#pragma unroll
  for (int i = 0; i < 8; ++i) { v[i] = __expf(v[i] - mx); s += v[i]; }
  float inv = 1.0f / block_sum(s);
#pragma unroll
  for (int i = 0; i < 8; ++i) out[base + tid + i * 256] = (__bf16)(v[i] * inv);
}

// ---------------------------------------------------------------------------
extern "C" void kernel_launch(void* const* d_in, const int* in_sizes, int n_in,
                              void* d_out, int out_size, void* d_ws, size_t ws_size,
                              hipStream_t stream) {
  (void)in_sizes; (void)n_in; (void)out_size; (void)ws_size;
  const float* x    = (const float*)d_in[0];
  const float* ln_g = (const float*)d_in[1];
  const float* ln_b = (const float*)d_in[2];
  const float* Wq   = (const float*)d_in[3];
  const float* bq   = (const float*)d_in[4];
  const float* Wk   = (const float*)d_in[5];
  const float* bk   = (const float*)d_in[6];
  const float* Wv   = (const float*)d_in[7];
  const float* bv   = (const float*)d_in[8];
  const float* Wp   = (const float*)d_in[9];
  const float* Wo   = (const float*)d_in[10];
  const float* bo   = (const float*)d_in[11];
  const float* ub   = (const float*)d_in[12];
  const float* vb   = (const float*)d_in[13];

  const int BS = BATCH * S_LEN;  // 4096
  size_t off = 0;
  auto alloc = [&](size_t bytes) -> void* {
    void* p = (char*)d_ws + off;
    off += (bytes + 255) & ~(size_t)255;
    return p;
  };
  __bf16* xn   = (__bf16*)alloc((size_t)BS * DMODEL * 2);
  __bf16* peb  = (__bf16*)alloc((size_t)S_LEN * DMODEL * 2);
  __bf16* Wqt  = (__bf16*)alloc((size_t)DMODEL * DMODEL * 2);
  __bf16* Wkt  = (__bf16*)alloc((size_t)DMODEL * DMODEL * 2);
  __bf16* Wvt  = (__bf16*)alloc((size_t)DMODEL * DMODEL * 2);
  __bf16* Wpt  = (__bf16*)alloc((size_t)DMODEL * DMODEL * 2);
  __bf16* Wot  = (__bf16*)alloc((size_t)DMODEL * DMODEL * 2);
  float*  bqu  = (float*)alloc((size_t)DMODEL * 4);
  float*  bqv  = (float*)alloc((size_t)DMODEL * 4);
  __bf16* QU   = (__bf16*)alloc((size_t)BS * DMODEL * 2);
  __bf16* QV   = (__bf16*)alloc((size_t)BS * DMODEL * 2);
  __bf16* Kb   = (__bf16*)alloc((size_t)BS * DMODEL * 2);
  __bf16* Vb_  = (__bf16*)alloc((size_t)BS * DMODEL * 2);
  __bf16* Pb   = (__bf16*)alloc((size_t)S_LEN * DMODEL * 2);
  __bf16* vtb  = (__bf16*)alloc((size_t)BS * DMODEL * 2);
  __bf16* hb   = (__bf16*)alloc((size_t)BS * DMODEL * 2);
  float*  posb = (float*)alloc((size_t)S_LEN * S_LEN * 4);
  float*  scob = (float*)alloc((size_t)S_LEN * S_LEN * 4);
  __bf16* attnb= (__bf16*)alloc((size_t)S_LEN * S_LEN * 2);

  dim3 blk(256);

  layernorm_kernel<<<BS, 256, 0, stream>>>(x, ln_g, ln_b, xn);
  pe_kernel<<<S_LEN, 256, 0, stream>>>(peb);

  int wgrid = (DMODEL * DMODEL + 255) / 256;
  transpose_w_kernel<<<wgrid, 256, 0, stream>>>(Wq, Wqt, DMODEL, DMODEL);
  transpose_w_kernel<<<wgrid, 256, 0, stream>>>(Wk, Wkt, DMODEL, DMODEL);
  transpose_w_kernel<<<wgrid, 256, 0, stream>>>(Wv, Wvt, DMODEL, DMODEL);
  transpose_w_kernel<<<wgrid, 256, 0, stream>>>(Wp, Wpt, DMODEL, DMODEL);
  transpose_w_kernel<<<wgrid, 256, 0, stream>>>(Wo, Wot, DMODEL, DMODEL);
  make_bias_kernel<<<(DMODEL + 255) / 256, 256, 0, stream>>>(bq, ub, vb, bqu, bqv);

  // projections (bf16 out)
  dim3 gproj((DMODEL + 127) / 128, (BS + 127) / 128);
  gemm_nt_kernel<<<gproj, blk, 0, stream>>>(xn, DMODEL, Wqt, DMODEL, BS, DMODEL, DMODEL,
                                            bqu, nullptr, 0, 1.0f, nullptr, QU, DMODEL);
  gemm_nt_kernel<<<gproj, blk, 0, stream>>>(xn, DMODEL, Wqt, DMODEL, BS, DMODEL, DMODEL,
                                            bqv, nullptr, 0, 1.0f, nullptr, QV, DMODEL);
  gemm_nt_kernel<<<gproj, blk, 0, stream>>>(xn, DMODEL, Wkt, DMODEL, BS, DMODEL, DMODEL,
                                            bk, nullptr, 0, 1.0f, nullptr, Kb, DMODEL);
  gemm_nt_kernel<<<gproj, blk, 0, stream>>>(xn, DMODEL, Wvt, DMODEL, BS, DMODEL, DMODEL,
                                            bv, nullptr, 0, 1.0f, nullptr, Vb_, DMODEL);
  dim3 gpe((DMODEL + 127) / 128, (S_LEN + 127) / 128);
  gemm_nt_kernel<<<gpe, blk, 0, stream>>>(peb, DMODEL, Wpt, DMODEL, S_LEN, DMODEL, DMODEL,
                                          nullptr, nullptr, 0, 1.0f, nullptr, Pb, DMODEL);
  vtrans_kernel<<<(BS * DMODEL + 255) / 256, 256, 0, stream>>>(Vb_, vtb);

  // per-(batch, head) attention: pos GEMM -> score GEMM (+shift,scale) ->
  // softmax -> attn @ V
  dim3 gsc((S_LEN + 127) / 128, (S_LEN + 127) / 128);   // 16 x 16
  dim3 ghd((DHEAD + 127) / 128, (S_LEN + 127) / 128);   //  1 x 16
  const float scale = 0.125f;                           // dh^-0.5, dh=64
  for (int b = 0; b < BATCH; ++b) {
    for (int h = 0; h < NHEAD; ++h) {
      const __bf16* quo = QU + (size_t)b * S_LEN * DMODEL + h * DHEAD;
      const __bf16* qvo = QV + (size_t)b * S_LEN * DMODEL + h * DHEAD;
      const __bf16* kbo = Kb + (size_t)b * S_LEN * DMODEL + h * DHEAD;
      const __bf16* pho = Pb + h * DHEAD;
      const __bf16* vto = vtb + (size_t)(b * NHEAD + h) * DHEAD * S_LEN;
      __bf16*       hbo = hb + (size_t)b * S_LEN * DMODEL + h * DHEAD;

      gemm_nt_kernel<<<gsc, blk, 0, stream>>>(qvo, DMODEL, pho, DMODEL,
                                              S_LEN, S_LEN, DHEAD,
                                              nullptr, nullptr, 0, 1.0f,
                                              posb, nullptr, S_LEN);
      gemm_nt_kernel<<<gsc, blk, 0, stream>>>(quo, DMODEL, kbo, DMODEL,
                                              S_LEN, S_LEN, DHEAD,
                                              nullptr, posb, S_LEN, scale,
                                              scob, nullptr, S_LEN);
      softmax_kernel<<<S_LEN, 256, 0, stream>>>(scob, attnb);
      gemm_nt_kernel<<<ghd, blk, 0, stream>>>(attnb, S_LEN, vto, S_LEN,
                                              S_LEN, DHEAD, S_LEN,
                                              nullptr, nullptr, 0, 1.0f,
                                              nullptr, hbo, DMODEL);
    }
  }

  // output projection: fp32 result straight into d_out
  gemm_nt_kernel<<<gproj, blk, 0, stream>>>(hb, DMODEL, Wot, DMODEL, BS, DMODEL, DMODEL,
                                            bo, nullptr, 0, 1.0f,
                                            (float*)d_out, nullptr, DMODEL);
}
